// BiLSTM_CRF_NER_13950053777779
// MI455X (gfx1250) — matrix operand order repaired and checked by reading, not compiled, not run on hardware
//
#include <hip/hip_runtime.h>
#include <hip/hip_bf16.h>

typedef __bf16 bf16_t;
typedef __attribute__((ext_vector_type(16))) __bf16 v16bf;
typedef __attribute__((ext_vector_type(8)))  float  v8f;
typedef __attribute__((ext_vector_type(4)))  unsigned int v4u;

#define BB   64
#define SS   512
#define DD   768
#define HH   512
#define TT   21
#define G4H  2048   // 4*H
#define H2   1024   // 2*H

// ---------------------------------------------------------------------------
// WMMA fragment helpers (layouts per CDNA5 ISA 7.12.2, 16-bit A 16x32 / B 32x16)
// ---------------------------------------------------------------------------
union ABits { v16bf v; v4u u[2]; };

// A fragment: p already resolved to (row, kb) for this lane; needs p[0..7] and p[16..23]
__device__ __forceinline__ v16bf a_frag_bf16(const bf16_t* p) {
    ABits f;
    f.u[0] = *(const v4u*)(p);
    f.u[1] = *(const v4u*)(p + 16);
    return f.v;
}

// B fragment: p resolved to (col-row-of-W, kb) for this lane; needs p[0..7] and p[16..23]
__device__ __forceinline__ v16bf b_frag_bf16(const bf16_t* p) {
    ABits f;
    f.u[0] = *(const v4u*)(p);
    f.u[1] = *(const v4u*)(p + 16);
    return f.v;
}

__device__ __forceinline__ v8f wmma_bf16(v16bf a, v16bf b, v8f c) {
    return __builtin_amdgcn_wmma_f32_16x16x32_bf16(false, a, false, b, (short)0, c, false, false);
}

__device__ __forceinline__ float sigmoidf(float x) {
    return 1.0f / (1.0f + __expf(-x));
}

// ---------------------------------------------------------------------------
// 0) Convert weights + x to bf16 in workspace (grid-stride)
// ---------------------------------------------------------------------------
__global__ __launch_bounds__(256) void convert_kernel(
    const float* __restrict__ x,
    const float* __restrict__ wihf, const float* __restrict__ whhf,
    const float* __restrict__ wihb, const float* __restrict__ whhb,
    bf16_t* __restrict__ x_bf, bf16_t* __restrict__ wih_bf,
    bf16_t* __restrict__ whh_bf) {
    const int NX   = BB * SS * DD;   // 25,165,824
    const int NWIH = G4H * DD;       //  1,572,864
    const int NWHH = G4H * HH;       //  1,048,576
    const int stride = gridDim.x * 256;
    for (int i = blockIdx.x * 256 + threadIdx.x; i < NX; i += stride) {
        x_bf[i] = (bf16_t)x[i];
        if (i < NWIH) {
            wih_bf[i]        = (bf16_t)wihf[i];
            wih_bf[NWIH + i] = (bf16_t)wihb[i];
        }
        if (i < NWHH) {
            whh_bf[i]        = (bf16_t)whhf[i];
            whh_bf[NWHH + i] = (bf16_t)whhb[i];
        }
    }
}

// ---------------------------------------------------------------------------
// 1) Input projection GEMM: xg[dir][s*64+b][4H] = x[b][s][:] @ Wih^T + bias
//    M = S*B = 32768, N = 4H = 2048, K = D = 768, bf16 WMMA, f32 accum.
//    grid = (M/16, N/1024, 2dirs), block = 256 (8 waves, 8 col-tiles each)
// ---------------------------------------------------------------------------
__global__ __launch_bounds__(256) void xproj_kernel(
    const bf16_t* __restrict__ x_bf, const bf16_t* __restrict__ wih_bf,
    const float* __restrict__ bias_f, const float* __restrict__ bias_b,
    float* __restrict__ xg_all) {
    const int dir = blockIdx.z;
    const bf16_t* W    = wih_bf + (size_t)dir * (G4H * DD);
    const float*  bias = dir ? bias_b : bias_f;
    float*        out  = xg_all + (size_t)dir * ((size_t)SS * BB * G4H);

    const int wave = threadIdx.x >> 5;
    const int lane = threadIdx.x & 31;
    const int half = lane >> 4;       // 0: lanes 0-15, 1: lanes 16-31
    const int colL = lane & 15;

    const int m0 = blockIdx.x * 16;   // row tile: m = s*64+b, same s across tile
    const int s  = m0 >> 6;
    const int b0 = m0 & 63;
    const int n0 = blockIdx.y * 1024;

    // per-lane A base: row = b0+colL (batch), K along D, row stride = S*D
    const bf16_t* abase = x_bf + ((size_t)(b0 + colL) * SS + s) * DD + (half << 3);
    // per-lane B base: output row n (Wih row), contiguous along D
    const bf16_t* bbase = W + (size_t)(n0 + wave * 16 + colL) * DD + (half << 3);

    v8f acc[8];
#pragma unroll
    for (int j = 0; j < 8; ++j) acc[j] = v8f{0,0,0,0,0,0,0,0};

    for (int kt = 0; kt < DD / 32; ++kt) {
        const int k0 = kt * 32;
        v16bf A = a_frag_bf16(abase + k0);
#pragma unroll
        for (int j = 0; j < 8; ++j) {
            v16bf Bf = b_frag_bf16(bbase + (size_t)j * 128 * DD + k0);
            acc[j] = wmma_bf16(A, Bf, acc[j]);
        }
    }

#pragma unroll
    for (int j = 0; j < 8; ++j) {
        const int n = n0 + j * 128 + wave * 16 + colL;
        const float bv = bias[n];
#pragma unroll
        for (int r = 0; r < 8; ++r) {
            const int m = m0 + r + (half << 3);   // C layout: VGPR r -> M = r (+8 hi half)
            out[(size_t)m * G4H + n] = acc[j][r] + bv;
        }
    }
}

// ---------------------------------------------------------------------------
// 2) Persistent recurrent LSTM kernel. grid = 2 (dir), block = 1024 (32 waves).
//    h double-buffered in LDS (2 x 64x512 bf16); cell state lives in VGPRs.
//    Wave w: row-block mrow = w&3 (16 of 64 batch rows), hset = w>>2 (0..7);
//    handles 4 combos hblock = hset + 8c, each combo = 4 gate tiles (i,f,g,o).
//    Inner K loop is software-pipelined one k-step ahead; next timestep's xg
//    slab is prefetched into L2 (global_prefetch_b8) while WMMAs run.
// ---------------------------------------------------------------------------
__global__ __launch_bounds__(1024) void lstm_kernel(
    const float* __restrict__ xg_all, const bf16_t* __restrict__ whh_bf,
    bf16_t* __restrict__ hs) {
    extern __shared__ char smem[];
    bf16_t* hbuf0 = (bf16_t*)smem;            // 64*512 bf16 = 64KB
    bf16_t* hbuf1 = hbuf0 + BB * HH;

    const int dir = blockIdx.x;
    const float*  xg  = xg_all + (size_t)dir * ((size_t)SS * BB * G4H);
    const bf16_t* Whh = whh_bf + (size_t)dir * (G4H * HH);

    const int tid  = threadIdx.x;
    const int wave = tid >> 5;
    const int lane = tid & 31;
    const int half = lane >> 4;
    const int colL = lane & 15;
    const int mrow = wave & 3;
    const int hset = wave >> 2;
    const int m0   = mrow * 16;

    for (int i = tid; i < BB * HH; i += 1024) hbuf0[i] = (bf16_t)0.0f;
    __syncthreads();

    float cell[4][8];
#pragma unroll
    for (int c = 0; c < 4; ++c)
#pragma unroll
        for (int r = 0; r < 8; ++r) cell[c][r] = 0.0f;

    bf16_t* cur = hbuf0;
    bf16_t* nxt = hbuf1;

    for (int step = 0; step < SS; ++step) {
        const int t = dir ? (SS - 1 - step) : step;

        // Prefetch next timestep's xg slab into L2: across the 32 waves the
        // (hset, half, colL) decomposition covers all 64 rows x 2048 cols.
        if (step + 1 < SS) {
            const int t2 = dir ? (SS - 2 - step) : (step + 1);
            const float* pfrow = xg + ((size_t)t2 * BB + m0 + colL) * G4H
                                    + hset * 64 + half * 32;
#pragma unroll
            for (int g = 0; g < 4; ++g)
                __builtin_prefetch(pfrow + g * HH, 0, 0);
        }

#pragma unroll
        for (int c = 0; c < 4; ++c) {
            const int hb   = hset + c * 8;       // hidden block 0..31
            const int hcol = hb * 16 + colL;     // hidden column this lane owns

            v8f acc0 = v8f{0,0,0,0,0,0,0,0};
            v8f acc1 = v8f{0,0,0,0,0,0,0,0};
            v8f acc2 = v8f{0,0,0,0,0,0,0,0};
            v8f acc3 = v8f{0,0,0,0,0,0,0,0};

            const bf16_t* ab = cur + (m0 + colL) * HH + (half << 3);
            const bf16_t* bb = Whh + (size_t)hcol * HH + (half << 3);

            // 1-deep software pipeline over the K dimension
            v16bf A  = a_frag_bf16(ab);
            v16bf B0 = b_frag_bf16(bb);
            v16bf B1 = b_frag_bf16(bb + (size_t)1 * HH * HH);
            v16bf B2 = b_frag_bf16(bb + (size_t)2 * HH * HH);
            v16bf B3 = b_frag_bf16(bb + (size_t)3 * HH * HH);
#pragma unroll
            for (int kt = 0; kt < HH / 32; ++kt) {
                const int kn = (kt + 1 < HH / 32) ? (kt + 1) * 32 : kt * 32;
                v16bf An  = a_frag_bf16(ab + kn);
                v16bf B0n = b_frag_bf16(bb + kn);
                v16bf B1n = b_frag_bf16(bb + (size_t)1 * HH * HH + kn);
                v16bf B2n = b_frag_bf16(bb + (size_t)2 * HH * HH + kn);
                v16bf B3n = b_frag_bf16(bb + (size_t)3 * HH * HH + kn);
                acc0 = wmma_bf16(A, B0, acc0);
                acc1 = wmma_bf16(A, B1, acc1);
                acc2 = wmma_bf16(A, B2, acc2);
                acc3 = wmma_bf16(A, B3, acc3);
                A = An; B0 = B0n; B1 = B1n; B2 = B2n; B3 = B3n;
            }

            const float* xgt = xg + (size_t)t * BB * G4H;
#pragma unroll
            for (int r = 0; r < 8; ++r) {
                const int m = m0 + r + (half << 3);
                const float* xrow = xgt + (size_t)m * G4H;
                const float gi = acc0[r] + xrow[hcol];
                const float gf = acc1[r] + xrow[HH + hcol];
                const float gg = acc2[r] + xrow[2 * HH + hcol];
                const float go = acc3[r] + xrow[3 * HH + hcol];
                const float cn = sigmoidf(gf) * cell[c][r] + sigmoidf(gi) * tanhf(gg);
                cell[c][r] = cn;
                const float h = sigmoidf(go) * tanhf(cn);
                nxt[m * HH + hcol] = (bf16_t)h;
                hs[((size_t)t * BB + m) * H2 + dir * HH + hcol] = (bf16_t)h;
            }
        }
        __syncthreads();
        bf16_t* tmp = cur; cur = nxt; nxt = tmp;
    }
}

// ---------------------------------------------------------------------------
// 3) Emissions: emis[s*64+b][T] = hs[s*64+b][:] @ W_out^T + b_out  (T = 21)
// ---------------------------------------------------------------------------
__global__ __launch_bounds__(32) void emis_kernel(
    const bf16_t* __restrict__ hs, const float* __restrict__ W_out,
    const float* __restrict__ b_out, float* __restrict__ emis) {
    const int sb  = blockIdx.x;
    const int tid = threadIdx.x;
    __shared__ float hrow[H2];
    const bf16_t* hp = hs + (size_t)sb * H2;
    for (int i = tid; i < H2; i += 32) hrow[i] = (float)hp[i];
    __syncthreads();
    if (tid < TT) {
        float sum = b_out[tid];
        const float* wr = W_out + (size_t)tid * H2;
        for (int k = 0; k < H2; ++k) sum += hrow[k] * wr[k];
        emis[(size_t)sb * TT + tid] = sum;
    }
}

// ---------------------------------------------------------------------------
// 4) Viterbi decode: one block per batch element, lane t owns tag t.
// ---------------------------------------------------------------------------
__global__ __launch_bounds__(32) void viterbi_kernel(
    const float* __restrict__ emis, const int* __restrict__ mask,
    const float* __restrict__ trans, const float* __restrict__ startt,
    const float* __restrict__ endt, int* __restrict__ hist,
    int* __restrict__ tags) {
    const int b = blockIdx.x;
    const int t = threadIdx.x;
    __shared__ float tr[TT * TT];
    __shared__ float sc[TT];
    for (int i = t; i < TT * TT; i += 32) tr[i] = trans[i];
    if (t < TT) sc[t] = startt[t] + emis[(size_t)b * TT + t];
    __syncthreads();

    for (int s = 1; s < SS; ++s) {
        float best = -3.4e38f;
        int   bi   = 0;
        float ns   = 0.0f;
        if (t < TT) {
            for (int tp = 0; tp < TT; ++tp) {
                const float v = sc[tp] + tr[tp * TT + t];
                if (v > best) { best = v; bi = tp; }
            }
            ns = best + emis[((size_t)s * BB + b) * TT + t];
        }
        const int mk = mask[b * SS + s];
        __syncthreads();   // all lanes done reading sc before overwrite
        if (t < TT) {
            if (mk) sc[t] = ns;
            hist[((size_t)(s - 1) * BB + b) * TT + t] = mk ? bi : 0;
        }
        __syncthreads();
    }

    if (t < TT) sc[t] += endt[t];
    __syncthreads();
    if (t == 0) {
        float best = -3.4e38f;
        int   tag  = 0;
        for (int i = 0; i < TT; ++i)
            if (sc[i] > best) { best = sc[i]; tag = i; }
        tags[b * SS + (SS - 1)] = tag;
        for (int s = SS - 2; s >= 0; --s) {
            tag = hist[((size_t)s * BB + b) * TT + tag];
            tags[b * SS + s] = tag;
        }
    }
}

// ---------------------------------------------------------------------------
// Launcher
// ---------------------------------------------------------------------------
extern "C" void kernel_launch(void* const* d_in, const int* in_sizes, int n_in,
                              void* d_out, int out_size, void* d_ws, size_t ws_size,
                              hipStream_t stream) {
    const float* x      = (const float*)d_in[0];
    const int*   mask   = (const int*)d_in[1];
    const float* Wih_f  = (const float*)d_in[2];
    const float* Whh_f  = (const float*)d_in[3];
    const float* b_f    = (const float*)d_in[4];
    const float* Wih_b  = (const float*)d_in[5];
    const float* Whh_b  = (const float*)d_in[6];
    const float* b_b    = (const float*)d_in[7];
    const float* W_out  = (const float*)d_in[8];
    const float* b_out  = (const float*)d_in[9];
    const float* trans  = (const float*)d_in[10];
    const float* startt = (const float*)d_in[11];
    const float* endt   = (const float*)d_in[12];
    int* out = (int*)d_out;

    char* w = (char*)d_ws;
    const size_t SZ_XG   = (size_t)2 * SS * BB * G4H * sizeof(float);   // 512 MB
    const size_t SZ_HS   = (size_t)SS * BB * H2 * sizeof(bf16_t);       //  64 MB
    const size_t SZ_XBF  = (size_t)BB * SS * DD * sizeof(bf16_t);       //  48 MB
    const size_t SZ_WIH  = (size_t)2 * G4H * DD * sizeof(bf16_t);
    const size_t SZ_WHH  = (size_t)2 * G4H * HH * sizeof(bf16_t);
    const size_t SZ_EMIS = (size_t)SS * BB * TT * sizeof(float);
    float*  xg     = (float*)(w);
    bf16_t* hs     = (bf16_t*)(w + SZ_XG);
    bf16_t* x_bf   = (bf16_t*)(w + SZ_XG + SZ_HS);
    bf16_t* wih_bf = (bf16_t*)(w + SZ_XG + SZ_HS + SZ_XBF);
    bf16_t* whh_bf = (bf16_t*)(w + SZ_XG + SZ_HS + SZ_XBF + SZ_WIH);
    float*  emis   = (float*)(w + SZ_XG + SZ_HS + SZ_XBF + SZ_WIH + SZ_WHH);
    int*    hist   = (int*)(w + SZ_XG + SZ_HS + SZ_XBF + SZ_WIH + SZ_WHH + SZ_EMIS);

    // 0) bf16 copies of x and weights
    convert_kernel<<<8192, 256, 0, stream>>>(x, Wih_f, Whh_f, Wih_b, Whh_b,
                                             x_bf, wih_bf, whh_bf);

    // 1) input projections (both directions)
    dim3 g1((SS * BB) / 16, G4H / 1024, 2);
    xproj_kernel<<<g1, 256, 0, stream>>>(x_bf, wih_bf, b_f, b_b, xg);

    // 2) recurrent bi-LSTM (one persistent workgroup per direction)
    lstm_kernel<<<2, 1024, 2 * BB * HH * sizeof(bf16_t), stream>>>(xg, whh_bf, hs);

    // 3) emissions
    emis_kernel<<<SS * BB, 32, 0, stream>>>(hs, W_out, b_out, emis);

    // 4) Viterbi
    viterbi_kernel<<<BB, 32, 0, stream>>>(emis, mask, trans, startt, endt, hist, out);
}
